// LongContextAttentionQKVPacked_91276644974767
// MI455X (gfx1250) — matrix operand order repaired and checked by reading, not compiled, not run on hardware
//
#include <hip/hip_runtime.h>
#include <hip/hip_bf16.h>

typedef __attribute__((ext_vector_type(16))) _Float16 v16h;
typedef __attribute__((ext_vector_type(8)))  _Float16 v8h;
typedef __attribute__((ext_vector_type(4)))  _Float16 v4h;
typedef __attribute__((ext_vector_type(8)))  float    v8f;

#define WMMA_F16(a, b, c) \
  __builtin_amdgcn_wmma_f32_16x16x32_f16(false, (a), false, (b), (short)0, (c), false, false)

static constexpr int Bb = 2, Ss = 2048, Hh = 16, Dd = 128;
static constexpr int BM = 64;   // query rows per workgroup (16 per wave, 4 waves)
static constexpr int BN = 32;   // key/value rows per iteration
static constexpr float SCALE  = 0.08838834764831845f;  // 1/sqrt(128)
static constexpr float LOG2E  = 1.4426950408889634f;
static constexpr float NEGF   = -1.0e30f;

// Raw hardware v_exp_f32 (args are always <= 0 here; flush-to-zero underflow is
// exactly what softmax wants, so skip the OCML ldexp range-handling).
__device__ __forceinline__ float fast_exp2(float x) {
  return __builtin_amdgcn_exp2f(x);
}

// ---- DPP 16-lane butterfly reductions (no LDS, VALU only) -------------------
// xor1 = quad_perm[1,0,3,2] (0xB1), xor2 = quad_perm[2,3,0,1] (0x4E),
// xor7 = ROW_HALF_MIRROR (0x141),  xor15 = ROW_MIRROR (0x140).
// span{1,2,7,15} = full 16-lane group; DPP rows never cross the half-wave.
template <int CTRL>
__device__ __forceinline__ float dpp_mov_f32(float x) {
  return __int_as_float(__builtin_amdgcn_update_dpp(
      0, __float_as_int(x), CTRL, 0xF, 0xF, true));
}
__device__ __forceinline__ float red16_max(float x) {
  x = fmaxf(x, dpp_mov_f32<0x0B1>(x));
  x = fmaxf(x, dpp_mov_f32<0x04E>(x));
  x = fmaxf(x, dpp_mov_f32<0x141>(x));
  x = fmaxf(x, dpp_mov_f32<0x140>(x));
  return x;
}
__device__ __forceinline__ float red16_sum(float x) {
  x += dpp_mov_f32<0x0B1>(x);
  x += dpp_mov_f32<0x04E>(x);
  x += dpp_mov_f32<0x141>(x);
  x += dpp_mov_f32<0x140>(x);
  return x;
}

__launch_bounds__(128, 2)
__global__ void fa_fwd_kernel(const float* __restrict__ qkv, float* __restrict__ out)
{
  // LDS: K tile (row-major f16), V tile transposed (d-major f16), per-wave P scratch
  __shared__ _Float16 Kl[BN * Dd];        // 8 KB : Kl[key_local*128 + d]
  __shared__ _Float16 Vt[Dd * BN];        // 8 KB : Vt[d*32 + key_local]
  __shared__ _Float16 Pb[4][16][BN];      // 4 KB : per-wave P tile [row][key_local]

  const int nmb  = Ss / BM;
  const int mblk = blockIdx.x % nmb;
  const int bh   = blockIdx.x / nmb;
  const int h    = bh % Hh;
  const int b    = bh / Hh;
  const int qm0  = mblk * BM;

  const int tid  = threadIdx.x;
  const int w    = tid >> 5;         // wave id 0..3
  const int lane = tid & 31;
  const int g    = (lane >> 4) & 1;  // half-wave group
  const int ln   = lane & 15;
  const int qrow0 = qm0 + 16 * w;    // this wave's 16 query rows

  const size_t rstride = (size_t)3 * Hh * Dd;       // floats between consecutive s
  const float* qb  = qkv + (size_t)b * Ss * rstride + (size_t)h * Dd;
  const float* kbp = qb + (size_t)Hh * Dd;          // c=1 plane
  const float* vbp = qb + (size_t)2 * Hh * Dd;      // c=2 plane

  // ---- Load Q tile (16x128) into 4 A-fragments (f16, pre-scaled) ----------
  // A k order used here (contiguous, same as B): lane group 0 (l<16): elems 0..15 -> K=0..15
  //                 lane group 1       : elems 0..15 -> K=16..31
  v16h aq[4];
  {
    const float* qrow = qb + (size_t)(qrow0 + ln) * rstride;
    #pragma unroll
    for (int c = 0; c < 4; ++c) {
      const int d0 = 32 * c + 16 * g;
      float4 x0 = *(const float4*)(qrow + d0);
      float4 x1 = *(const float4*)(qrow + d0 + 4);
      float4 y0 = *(const float4*)(qrow + d0 + 8);
      float4 y1 = *(const float4*)(qrow + d0 + 12);
      float t[16] = { x0.x, x0.y, x0.z, x0.w,  x1.x, x1.y, x1.z, x1.w,
                      y0.x, y0.y, y0.z, y0.w,  y1.x, y1.y, y1.z, y1.w };
      v16h a;
      #pragma unroll
      for (int i = 0; i < 16; ++i) a[i] = (_Float16)(t[i] * SCALE);
      aq[c] = a;
    }
  }

  // ---- Accumulators -------------------------------------------------------
  v8f o[8];
  #pragma unroll
  for (int dt = 0; dt < 8; ++dt) o[dt] = (v8f)0.0f;
  float m_[8], l_[8];
  #pragma unroll
  for (int r = 0; r < 8; ++r) { m_[r] = -3.0e38f; l_[r] = 0.0f; }

  const int kbend = qm0 / BN + 2;   // key blocks up to (and including) the diagonal

  for (int kb = 0; kb < kbend; ++kb) {
    const int key0 = kb * BN;

    __syncthreads();
    // ---- Cooperative staging: K -> Kl (f16 row-major), V -> Vt (f16 transposed)
    {
      const int row = tid >> 2;            // key_local 0..31
      const int cq  = (tid & 3) * 32;      // d quarter
      const float* kr = kbp + (size_t)(key0 + row) * rstride + cq;
      const float* vr = vbp + (size_t)(key0 + row) * rstride + cq;
      #pragma unroll
      for (int j = 0; j < 32; j += 4) {
        float4 kf = *(const float4*)(kr + j);
        v4h kh = { (_Float16)kf.x, (_Float16)kf.y, (_Float16)kf.z, (_Float16)kf.w };
        *(v4h*)&Kl[row * Dd + cq + j] = kh;
        float4 vf = *(const float4*)(vr + j);
        Vt[(cq + j + 0) * BN + row] = (_Float16)vf.x;
        Vt[(cq + j + 1) * BN + row] = (_Float16)vf.y;
        Vt[(cq + j + 2) * BN + row] = (_Float16)vf.z;
        Vt[(cq + j + 3) * BN + row] = (_Float16)vf.w;
      }
      if (kb + 1 < kbend) {               // prefetch next key block (global_prefetch_b8)
        __builtin_prefetch(kr + BN * rstride, 0, 1);
        __builtin_prefetch(vr + BN * rstride, 0, 1);
      }
    }
    __syncthreads();

    // Wave-uniform skip of fully-masked blocks (EXEC stays all-1s for WMMA)
    if (key0 > qrow0 + 15) continue;

    // ---- S = Q * K^T : two 16x16 tiles, 4 K-chunks each ---------------------
    v8f sc[2];
    sc[0] = (v8f)0.0f; sc[1] = (v8f)0.0f;
    #pragma unroll
    for (int t = 0; t < 2; ++t) {
      const int krl = 16 * t + ln;        // local key row this lane reads
      #pragma unroll
      for (int c = 0; c < 4; ++c) {
        // B uses the same k order as A: lane reads K[n][d0+16g .. d0+16g+15] contiguous
        v16h bk = *(const v16h*)&Kl[krl * Dd + 32 * c + 16 * g];
        sc[t] = WMMA_F16(aq[c], bk, sc[t]);
      }
    }

    // ---- Causal mask on diagonal blocks ------------------------------------
    if (key0 + BN - 1 > qrow0) {
      #pragma unroll
      for (int t = 0; t < 2; ++t) {
        const int kc = key0 + 16 * t + ln;
        #pragma unroll
        for (int r = 0; r < 8; ++r) {
          const int qr = qrow0 + r + 8 * g;
          if (kc > qr) sc[t][r] = NEGF;
        }
      }
    }

    // ---- Online softmax (DPP reductions, raw v_exp_f32) --------------------
    float mn[8], al[8];
    #pragma unroll
    for (int r = 0; r < 8; ++r) {
      float rm = red16_max(fmaxf(sc[0][r], sc[1][r]));
      mn[r] = fmaxf(m_[r], rm);
      al[r] = fast_exp2((m_[r] - mn[r]) * LOG2E);
      m_[r] = mn[r];
    }
    #pragma unroll
    for (int t = 0; t < 2; ++t)
      #pragma unroll
      for (int r = 0; r < 8; ++r)
        sc[t][r] = fast_exp2((sc[t][r] - mn[r]) * LOG2E);
    #pragma unroll
    for (int r = 0; r < 8; ++r) {
      float rsv = red16_sum(sc[0][r] + sc[1][r]);
      l_[r] = l_[r] * al[r] + rsv;
    }
    #pragma unroll
    for (int dt = 0; dt < 8; ++dt)
      #pragma unroll
      for (int r = 0; r < 8; ++r)
        o[dt][r] *= al[r];

    // ---- C-layout -> A-layout via per-wave LDS tile ------------------------
    #pragma unroll
    for (int t = 0; t < 2; ++t)
      #pragma unroll
      for (int r = 0; r < 8; ++r)
        Pb[w][r + 8 * g][16 * t + ln] = (_Float16)sc[t][r];

    v16h pa;
    {
      v8h p0 = *(const v8h*)&Pb[w][ln][16 * g];
      v8h p1 = *(const v8h*)&Pb[w][ln][16 * g + 8];
      #pragma unroll
      for (int i = 0; i < 8; ++i) { pa[i] = p0[i]; pa[i + 8] = p1[i]; }
    }

    // ---- O += P * V : 8 D-tiles --------------------------------------------
    #pragma unroll
    for (int dt = 0; dt < 8; ++dt) {
      // B element (k,n) = V[key0+k][16dt+n] = Vt[(16dt+n)*32 + k]; contiguous per lane
      v16h bv = *(const v16h*)&Vt[(16 * dt + ln) * BN + 16 * g];
      o[dt] = WMMA_F16(pa, bv, o[dt]);
    }
  }

  // ---- Epilogue: out[b,q,h,d] = O / l -------------------------------------
  float invl[8];
  #pragma unroll
  for (int r = 0; r < 8; ++r) invl[r] = 1.0f / l_[r];
  #pragma unroll
  for (int dt = 0; dt < 8; ++dt) {
    #pragma unroll
    for (int r = 0; r < 8; ++r) {
      const int qr = qrow0 + r + 8 * g;
      out[(((size_t)b * Ss + qr) * Hh + h) * Dd + 16 * dt + ln] = o[dt][r] * invl[r];
    }
  }
}

extern "C" void kernel_launch(void* const* d_in, const int* in_sizes, int n_in,
                              void* d_out, int out_size, void* d_ws, size_t ws_size,
                              hipStream_t stream) {
  (void)in_sizes; (void)n_in; (void)out_size; (void)d_ws; (void)ws_size;
  const float* qkv = (const float*)d_in[0];
  float* out = (float*)d_out;
  dim3 grid(Bb * Hh * (Ss / BM));   // 1024 workgroups
  dim3 block(128);                  // 4 waves
  fa_fwd_kernel<<<grid, block, 0, stream>>>(qkv, out);
}
